// FlowFaceCrossAttentionLayer_34282428956917
// MI455X (gfx1250) — matrix operand order repaired and checked by reading, not compiled, not run on hardware
//
#include <hip/hip_runtime.h>

typedef __attribute__((ext_vector_type(16))) _Float16 v16h;
typedef __attribute__((ext_vector_type(8)))  _Float16 v8h;
typedef __attribute__((ext_vector_type(4)))  _Float16 v4h;
typedef __attribute__((ext_vector_type(8)))  float    v8f;

constexpr int BB   = 4;
constexpr int SS   = 4096;
constexpr int DD   = 512;
constexpr int MTOT = BB * SS;   // 16384 tokens

// ---------------------------------------------------------------------------
// CDNA5 async memory->LDS copy (ASYNCcnt-tracked, no VGPR round trip).
// dsaddr VGPR = LDS byte address, vaddr = 64-bit global address (GV mode).
// ---------------------------------------------------------------------------
__device__ __forceinline__ void async_ld_b128(void* lds, const void* gptr) {
  unsigned int l = (unsigned int)(unsigned long long)lds;
  unsigned long long g = (unsigned long long)gptr;
  asm volatile("global_load_async_to_lds_b128 %0, %1, off"
               :: "v"(l), "v"(g) : "memory");
}
__device__ __forceinline__ void wait_async0() {
  asm volatile("s_wait_asynccnt 0x0" ::: "memory");
}

// ---------------------------------------------------------------------------
// WMMA fragment loaders (wave32 layouts per CDNA5 ISA 7.12.2)
// A 16x32 f16: lane l = M row; halfs e0..7 -> K = hi*8+e, e8..15 -> K = 16+hi*8+(e-8)
// B 32x16 f16: lane l = N col; halfs e0..7 -> K = hi*8+e, e8..15 -> K = 16+hi*8+(e-8)
// C/D 16x16 f32: vgpr v -> row = v + 8*hi, col = l
// ---------------------------------------------------------------------------
__device__ __forceinline__ v16h load_afrag(const _Float16* p) {
  union { v16h v; v8h h[2]; } u;
  u.h[0] = *(const v8h*)(p);
  u.h[1] = *(const v8h*)(p + 16);
  return u.v;
}
__device__ __forceinline__ v16h load_bfrag(const _Float16* p) {
  union { v16h v; v8h h[2]; } u;
  u.h[0] = *(const v8h*)(p);
  u.h[1] = *(const v8h*)(p + 16);
  return u.v;
}

// ---------------------------------------------------------------------------
// f32 -> f16 conversion
// ---------------------------------------------------------------------------
__global__ void cvt_f32_to_f16(const float* __restrict__ in,
                               _Float16* __restrict__ out, int n) {
  int i = (blockIdx.x * blockDim.x + threadIdx.x) * 4;
  if (i + 3 < n) {
    float4 v = *(const float4*)(in + i);
    v4h o;
    o[0] = (_Float16)v.x; o[1] = (_Float16)v.y;
    o[2] = (_Float16)v.z; o[3] = (_Float16)v.w;
    *(v4h*)(out + i) = o;
  }
}

// ---------------------------------------------------------------------------
// C[M,512] = A[M,512] @ W[512,512]^T + bias  (torch Linear semantics)
// Block: 128(M) x 64(N), 8 waves; double-buffered async LDS staging.
// ---------------------------------------------------------------------------
template <typename OT, bool TRANS_OUT>
__global__ __launch_bounds__(256) void gemm_bias_512(
    const _Float16* __restrict__ A,    // [MTOT, 512]
    const _Float16* __restrict__ W,    // [512, 512]
    const float*    __restrict__ bias, // [512]
    OT* __restrict__ C)
{
  __shared__ _Float16 As[2][128][40];   // +8 halfs pad, rows stay 16B aligned
  __shared__ _Float16 Ws[2][64][40];

  const int tid  = threadIdx.x;
  const int wave = tid >> 5;
  const int lane = tid & 31;
  const int l    = lane & 15;
  const int hi   = lane >> 4;
  const int m0   = blockIdx.y * 128;
  const int n0   = blockIdx.x * 64;

  // per-thread staging coordinates
  const int ar0 = (tid * 8) >> 5,            ad0 = (tid * 8) & 31;
  const int ar1 = ((256 + tid) * 8) >> 5,    ad1 = ((256 + tid) * 8) & 31;

  v8f acc[4] = {};

  // prologue: issue K-tile 0
  {
    const int kk0 = 0;
    async_ld_b128(&As[0][ar0][ad0], A + (size_t)(m0 + ar0) * DD + kk0 + ad0);
    async_ld_b128(&As[0][ar1][ad1], A + (size_t)(m0 + ar1) * DD + kk0 + ad1);
    async_ld_b128(&Ws[0][ar0][ad0], W + (size_t)(n0 + ar0) * DD + kk0 + ad0);
  }

  for (int kt = 0; kt < 16; ++kt) {
    const int cur = kt & 1;
    wait_async0();
    __syncthreads();
    if (kt < 15) {        // overlap next tile's DMA with this tile's WMMAs
      const int nkk = (kt + 1) * 32;
      async_ld_b128(&As[1 - cur][ar0][ad0], A + (size_t)(m0 + ar0) * DD + nkk + ad0);
      async_ld_b128(&As[1 - cur][ar1][ad1], A + (size_t)(m0 + ar1) * DD + nkk + ad1);
      async_ld_b128(&Ws[1 - cur][ar0][ad0], W + (size_t)(n0 + ar0) * DD + nkk + ad0);
    }
    v16h a = load_afrag(&As[cur][wave * 16 + l][hi * 8]);
#pragma unroll
    for (int j = 0; j < 4; ++j) {
      v16h bfr = load_bfrag(&Ws[cur][j * 16 + l][hi * 8]);
      acc[j] = __builtin_amdgcn_wmma_f32_16x16x32_f16(
          false, a, false, bfr, (short)0, acc[j], false, false);
    }
  }

#pragma unroll
  for (int j = 0; j < 4; ++j) {
    const int n  = n0 + j * 16 + l;
    const float bv = bias[n];
#pragma unroll
    for (int v = 0; v < 8; ++v) {
      const int m = m0 + wave * 16 + v + hi * 8;
      float val = acc[j][v] + bv;
      if constexpr (TRANS_OUT) {
        int bi = m >> 12;          // m / 4096
        int si = m & 4095;
        C[((size_t)bi * DD + n) * SS + si] = (OT)val;   // [B, D, S]
      } else {
        C[(size_t)m * DD + n] = (OT)val;                 // [B, S, D]
      }
    }
  }
}

// ---------------------------------------------------------------------------
// Fused flash attention: out = softmax(QK^T) @ KV + QV  (no 1/sqrt(d), faithful)
// One block = one batch x 64 query rows; key loop in tiles of 64.
// Q tile resident in LDS; K/V staged via double-buffered async DMA.
// ---------------------------------------------------------------------------
__global__ __launch_bounds__(256) void flash_attn_kernel(
    const _Float16* __restrict__ Q,    // [B,S,D]
    const _Float16* __restrict__ Km,   // [B,S,D]
    const _Float16* __restrict__ VT,   // [B,D,S]  (kv, transposed)
    const float*    __restrict__ QV,   // [B,S,D]  residual stream
    _Float16* __restrict__ Out)        // [B,S,D]
{
  __shared__ _Float16 Qs[64][520];
  __shared__ _Float16 Ks[2][64][136];
  __shared__ _Float16 Vs[2][128][72];
  __shared__ _Float16 Ps[64][72];
  __shared__ float row_max[64], row_sum[64];
  __shared__ float pmax[2][64], psum[2][64];

  const int tid  = threadIdx.x;
  const int wave = tid >> 5;
  const int lane = tid & 31;
  const int l    = lane & 15;
  const int hi   = lane >> 4;
  const int mt   = wave & 3;
  const int ch   = wave >> 2;
  const int bid  = blockIdx.x;
  const int b    = bid >> 6;           // / (S/64)
  const int q0   = (bid & 63) * 64;
  const size_t qbase = ((size_t)b * SS + q0) * DD;

  // stage full Q tile (64 x 512) via async DMA
#pragma unroll
  for (int it = 0; it < 16; ++it) {
    int idx = (it * 256 + tid) * 8;
    int r = idx >> 9, d = idx & 511;
    async_ld_b128(&Qs[r][d], Q + qbase + (size_t)r * DD + d);
  }
  if (tid < 64) { row_max[tid] = -3.0e38f; row_sum[tid] = 0.0f; }

  v8f oacc[16] = {};   // 16 rows x 256 cols per wave (128 VGPRs, wave32)

  for (int kt = 0; kt < 64; ++kt) {
    const int k0 = kt * 64;
    const size_t kbase = ((size_t)b * SS + k0) * DD;

    // ---- S = Q @ K^T, d-chunks of 128, double-buffered K staging ----
    v8f sacc[2] = {};
    {
      int idx = tid * 8;
      // issue chunk 0 (4 x b128 per thread)
#pragma unroll
      for (int it = 0; it < 4; ++it) {
        int id2 = idx + it * 2048;
        int r = id2 >> 7, d = id2 & 127;
        async_ld_b128(&Ks[0][r][d], Km + kbase + (size_t)r * DD + d);
      }
    }
    for (int c = 0; c < 4; ++c) {
      const int cur = c & 1;
      wait_async0();
      __syncthreads();
      if (c < 3) {
        int idx = tid * 8;
#pragma unroll
        for (int it = 0; it < 4; ++it) {
          int id2 = idx + it * 2048;
          int r = id2 >> 7, d = id2 & 127;
          async_ld_b128(&Ks[1 - cur][r][d],
                        Km + kbase + (size_t)r * DD + (c + 1) * 128 + d);
        }
      }
#pragma unroll
      for (int kk = 0; kk < 4; ++kk) {
        v16h a = load_afrag(&Qs[mt * 16 + l][c * 128 + kk * 32 + hi * 8]);
#pragma unroll
        for (int j = 0; j < 2; ++j) {
          int nt = 2 * ch + j;
          v16h bfr = load_bfrag(&Ks[cur][nt * 16 + l][kk * 32 + hi * 8]);
          sacc[j] = __builtin_amdgcn_wmma_f32_16x16x32_f16(
              false, a, false, bfr, (short)0, sacc[j], false, false);
        }
      }
    }
    __syncthreads();   // all waves done reading Ks before softmax/PV reuse

    // ---- online softmax ----
    float nm_a[8], sc_a[8];
#pragma unroll
    for (int v = 0; v < 8; ++v) {
      float lm = fmaxf(sacc[0][v], sacc[1][v]);
      lm = fmaxf(lm, __shfl_xor(lm, 1));
      lm = fmaxf(lm, __shfl_xor(lm, 2));
      lm = fmaxf(lm, __shfl_xor(lm, 4));
      lm = fmaxf(lm, __shfl_xor(lm, 8));
      if (l == 0) pmax[ch][mt * 16 + v + hi * 8] = lm;
    }
    __syncthreads();
#pragma unroll
    for (int v = 0; v < 8; ++v) {
      int row  = mt * 16 + v + hi * 8;
      float mo = row_max[row];
      float nm = fmaxf(mo, fmaxf(pmax[0][row], pmax[1][row]));
      float sc = __expf(mo - nm);
      nm_a[v] = nm; sc_a[v] = sc;
      float p0 = __expf(sacc[0][v] - nm);
      float p1 = __expf(sacc[1][v] - nm);
      Ps[row][(2 * ch) * 16 + l]     = (_Float16)p0;
      Ps[row][(2 * ch + 1) * 16 + l] = (_Float16)p1;
      float ls = p0 + p1;
      ls += __shfl_xor(ls, 1);
      ls += __shfl_xor(ls, 2);
      ls += __shfl_xor(ls, 4);
      ls += __shfl_xor(ls, 8);
      if (l == 0) psum[ch][row] = ls;
#pragma unroll
      for (int g = 0; g < 16; ++g) oacc[g][v] *= sc;   // rescale O
    }
    __syncthreads();
    if (ch == 0 && l == 0) {
#pragma unroll
      for (int v = 0; v < 8; ++v) {
        int row = mt * 16 + v + hi * 8;
        row_sum[row] = row_sum[row] * sc_a[v] + psum[0][row] + psum[1][row];
        row_max[row] = nm_a[v];
      }
    }
    __syncthreads();

    // ---- O += P @ V, double-buffered V staging from [B,D,S] ----
    v16h pa0 = load_afrag(&Ps[mt * 16 + l][0  + hi * 8]);
    v16h pa1 = load_afrag(&Ps[mt * 16 + l][32 + hi * 8]);
    {
      int idx = tid * 8;
#pragma unroll
      for (int it = 0; it < 4; ++it) {
        int id2 = idx + it * 2048;
        int r = id2 >> 6, d = id2 & 63;
        async_ld_b128(&Vs[0][r][d],
                      VT + ((size_t)b * DD + r) * SS + k0 + d);
      }
    }
    for (int c = 0; c < 4; ++c) {
      const int cur = c & 1;
      wait_async0();
      __syncthreads();
      if (c < 3) {
        int idx = tid * 8;
#pragma unroll
        for (int it = 0; it < 4; ++it) {
          int id2 = idx + it * 2048;
          int r = id2 >> 6, d = id2 & 63;
          async_ld_b128(&Vs[1 - cur][r][d],
                        VT + ((size_t)b * DD + (c + 1) * 128 + r) * SS + k0 + d);
        }
      }
#pragma unroll
      for (int t = 0; t < 4; ++t) {
        int nl = ch * 64 + t * 16;
        v16h b0 = load_bfrag(&Vs[cur][nl + l][0  + hi * 8]);
        v16h b1 = load_bfrag(&Vs[cur][nl + l][32 + hi * 8]);
        int g = c * 4 + t;
        oacc[g] = __builtin_amdgcn_wmma_f32_16x16x32_f16(
            false, pa0, false, b0, (short)0, oacc[g], false, false);
        oacc[g] = __builtin_amdgcn_wmma_f32_16x16x32_f16(
            false, pa1, false, b1, (short)0, oacc[g], false, false);
      }
    }
    __syncthreads();   // Vs/Ps stable until next iteration's overwrites
  }

  // ---- epilogue: O / l + QV, store f16 ----
#pragma unroll
  for (int g = 0; g < 16; ++g) {
    int c = g >> 2, t = g & 3;
    int col = c * 128 + ch * 64 + t * 16 + l;
#pragma unroll
    for (int v = 0; v < 8; ++v) {
      int row = mt * 16 + v + hi * 8;
      float val = oacc[g][v] / row_sum[row] +
                  QV[qbase + (size_t)row * DD + col];
      Out[qbase + (size_t)row * DD + col] = (_Float16)val;
    }
  }
}

// ---------------------------------------------------------------------------
// Host-side orchestration
// ---------------------------------------------------------------------------
extern "C" void kernel_launch(void* const* d_in, const int* in_sizes, int n_in,
                              void* d_out, int out_size, void* d_ws, size_t ws_size,
                              hipStream_t stream) {
  (void)in_sizes; (void)n_in; (void)out_size; (void)ws_size;
  const float* x   = (const float*)d_in[0];
  const float* y   = (const float*)d_in[1];
  const float* Wq  = (const float*)d_in[2];
  const float* bq  = (const float*)d_in[3];
  const float* Wqv = (const float*)d_in[4];
  const float* bqv = (const float*)d_in[5];
  const float* Wk  = (const float*)d_in[6];
  const float* bk  = (const float*)d_in[7];
  const float* Wkv = (const float*)d_in[8];
  const float* bkv = (const float*)d_in[9];
  const float* Wf  = (const float*)d_in[10];
  const float* bf  = (const float*)d_in[11];

  char* ws = (char*)d_ws;
  const size_t TOKH = (size_t)MTOT * DD * sizeof(_Float16);   // 16 MiB
  const size_t WH   = (size_t)DD * DD * sizeof(_Float16);     // 0.5 MiB
  _Float16* xh   = (_Float16*)ws; ws += TOKH;
  _Float16* yh   = (_Float16*)ws; ws += TOKH;
  _Float16* qh   = (_Float16*)ws; ws += TOKH;
  _Float16* kh   = (_Float16*)ws; ws += TOKH;
  _Float16* kvT  = (_Float16*)ws; ws += TOKH;
  _Float16* aoh  = (_Float16*)ws; ws += TOKH;
  float*    qvf  = (float*)ws;    ws += TOKH * 2;             // f32 residual
  _Float16* Wqh  = (_Float16*)ws; ws += WH;
  _Float16* Wqvh = (_Float16*)ws; ws += WH;
  _Float16* Wkh  = (_Float16*)ws; ws += WH;
  _Float16* Wkvh = (_Float16*)ws; ws += WH;
  _Float16* Wfh  = (_Float16*)ws; ws += WH;

  const int ntok = MTOT * DD;      // 8,388,608
  const int nwt  = DD * DD;        // 262,144
  cvt_f32_to_f16<<<ntok / 1024, 256, 0, stream>>>(x, xh, ntok);
  cvt_f32_to_f16<<<ntok / 1024, 256, 0, stream>>>(y, yh, ntok);
  cvt_f32_to_f16<<<nwt / 1024, 256, 0, stream>>>(Wq,  Wqh,  nwt);
  cvt_f32_to_f16<<<nwt / 1024, 256, 0, stream>>>(Wqv, Wqvh, nwt);
  cvt_f32_to_f16<<<nwt / 1024, 256, 0, stream>>>(Wk,  Wkh,  nwt);
  cvt_f32_to_f16<<<nwt / 1024, 256, 0, stream>>>(Wkv, Wkvh, nwt);
  cvt_f32_to_f16<<<nwt / 1024, 256, 0, stream>>>(Wf,  Wfh,  nwt);

  dim3 gg(DD / 64, MTOT / 128);    // (8, 128)
  gemm_bias_512<_Float16, false><<<gg, 256, 0, stream>>>(xh, Wqh,  bq,  qh);
  gemm_bias_512<float,    false><<<gg, 256, 0, stream>>>(xh, Wqvh, bqv, qvf);
  gemm_bias_512<_Float16, false><<<gg, 256, 0, stream>>>(yh, Wkh,  bk,  kh);
  gemm_bias_512<_Float16, true ><<<gg, 256, 0, stream>>>(yh, Wkvh, bkv, kvT);

  flash_attn_kernel<<<BB * (SS / 64), 256, 0, stream>>>(qh, kh, kvT, qvf, aoh);

  gemm_bias_512<float, false><<<gg, 256, 0, stream>>>(aoh, Wfh, bf, (float*)d_out);
}
